// ValueNet_47622597378172
// MI455X (gfx1250) — matrix operand order repaired and checked by reading, not compiled, not run on hardware
//
#include <hip/hip_runtime.h>
#include <cstddef>
#include <cstdint>

typedef __bf16 bf16;
typedef __attribute__((ext_vector_type(16))) __bf16 v16bf;
typedef __attribute__((ext_vector_type(8)))  float  v8f;

static constexpr int  kN    = 100000;   // multiple of 16 (6250 M-tiles)
static constexpr int  kE    = 1600000;
static constexpr int  kB    = 32;
static constexpr int  kDCTX = 135;

// ---------------------------------------------------------------- utilities
__device__ inline void edge_sd(const int* __restrict__ ei, long e, int E, int& s, int& d) {
  if (e < (long)E) { s = ei[e]; d = ei[(size_t)E + e]; }
  else             { s = (int)(e - E); d = s; }
}
__device__ inline unsigned f_flip(float x) {
  unsigned u = __float_as_uint(x);
  return (u & 0x80000000u) ? ~u : (u | 0x80000000u);
}
__device__ inline float f_unflip(unsigned u) {
  u = (u & 0x80000000u) ? (u & 0x7fffffffu) : ~u;
  return __uint_as_float(u);
}

__global__ void k_fill_f(float* __restrict__ p, float v, long n) {
  long i = (long)blockIdx.x * blockDim.x + threadIdx.x;
  if (i < n) p[i] = v;
}

// ---------------------------------------------- weight fp32 -> packed bf16
// dst layout: [Nout/16][Kpad/32][lane 0..31][16 bf16]  (WMMA B-fragment order)
// lane's element e holds K = kt*32 + (e<8 ? (lane>>4)*8 + e : 16 + (lane>>4)*8 + e-8), col = nt*16 + (lane&15)
// kmod: source row = k % kmod (row-replication for the l1 per-head projection)
__global__ void k_pack_w(const float* __restrict__ src, bf16* __restrict__ dst,
                         int K, int Nout, int Kpad, int kmod) {
  long i = (long)blockIdx.x * blockDim.x + threadIdx.x;
  long tot = (long)Kpad * Nout;
  if (i >= tot) return;
  int nkt = Kpad >> 5;
  int e    = (int)(i & 15);
  int lane = (int)((i >> 4) & 31);
  long rest = i >> 9;
  int kt = (int)(rest % nkt);
  int nt = (int)(rest / nkt);
  int k   = kt * 32 + (e < 8 ? (lane >> 4) * 8 + e : 16 + (lane >> 4) * 8 + (e - 8));
  int col = nt * 16 + (lane & 15);
  if (k < K) {
    int srck = k % kmod;
    dst[i] = (bf16)src[(size_t)srck * Nout + col];
  } else {
    dst[i] = (bf16)0.0f;
  }
}

// ---------------------------------------------------------- WMMA bf16 GEMM
// C[M,Nout] = act(A[M,lda](fp32, K-padded with zeros) @ W(packed bf16) + bias)
// one wave per 16x16 tile, 8 waves/block, fully unrolled K loop (NKT blocks).
// ACT: 0=none, 1=relu, 2=leaky(0.2). Compile-time -> straight-line epilogue.
template <int NKT, int ACT, bool HASBIAS>
__global__ void k_gemm_bf16(const float* __restrict__ A, int lda, int M,
                            const bf16* __restrict__ W, int Nout,
                            const float* __restrict__ bias, float* __restrict__ C) {
  const int lane  = threadIdx.x & 31;
  const int wave  = threadIdx.x >> 5;
  const int tileM = blockIdx.y * 8 + wave;
  if (tileM * 16 >= M) return;            // wave-uniform exit, EXEC stays full
  const int tileN = blockIdx.x;
  const int mrow  = tileM * 16 + (lane & 15);
  const int kh    = lane >> 4;
  const int bcol  = tileN * 16 + (lane & 15);

  const float* arow = A + (size_t)mrow * lda + kh * 8;            // 32B aligned
  const bf16*  wptr = W + ((size_t)tileN * NKT * 32 + lane) * 16; // 32B aligned

  v8f acc = {};
#pragma unroll
  for (int kt = 0; kt < NKT; ++kt) {
    const float4 a0 = *(const float4*)(arow + 0);
    const float4 a1 = *(const float4*)(arow + 4);
    const float4 a2 = *(const float4*)(arow + 16);
    const float4 a3 = *(const float4*)(arow + 20);
    v16bf af;
    af[0]  = (bf16)a0.x; af[1]  = (bf16)a0.y; af[2]  = (bf16)a0.z; af[3]  = (bf16)a0.w;
    af[4]  = (bf16)a1.x; af[5]  = (bf16)a1.y; af[6]  = (bf16)a1.z; af[7]  = (bf16)a1.w;
    af[8]  = (bf16)a2.x; af[9]  = (bf16)a2.y; af[10] = (bf16)a2.z; af[11] = (bf16)a2.w;
    af[12] = (bf16)a3.x; af[13] = (bf16)a3.y; af[14] = (bf16)a3.z; af[15] = (bf16)a3.w;
    const v16bf bfv = *(const v16bf*)wptr;
    acc = __builtin_amdgcn_wmma_f32_16x16x32_bf16(false, af, false, bfv,
                                                  (short)0, acc, false, false);
    arow += 32;
    wptr += 32 * 16;
  }
  const float bv = HASBIAS ? bias[bcol] : 0.f;
#pragma unroll
  for (int r = 0; r < 8; ++r) {
    int row = tileM * 16 + (lane >> 4) * 8 + r;  // C/D layout: VGPR r -> M row
    float v = acc[r] + bv;
    if (ACT == 1)      v = v > 0.f ? v : 0.f;        // relu
    else if (ACT == 2) v = v > 0.f ? v : 0.2f * v;   // leaky 0.2
    C[(size_t)row * Nout + bcol] = v;
  }
}

// ------------------------------------------------------------- GAT kernels
__global__ void k_build_xin(const float* __restrict__ x, const float* __restrict__ J,
                            const float* __restrict__ sz, float* __restrict__ xin, int N) {
  long i = (long)blockIdx.x * blockDim.x + threadIdx.x;
  if (i >= (long)N * 32) return;
  int n = (int)(i >> 5), j = (int)(i & 31);
  float v = 0.f;
  if (j < 6)       v = x[(size_t)n * 6 + j];
  else if (j == 6) v = J[n];
  else if (j == 7) v = sz[n];
  xin[i] = v;
}

__global__ void k_esed(const float* __restrict__ hp, const float* __restrict__ asrc,
                       const float* __restrict__ adst, float* __restrict__ es,
                       float* __restrict__ ed, int N) {
  long i = (long)blockIdx.x * blockDim.x + threadIdx.x;
  if (i >= (long)N * 4) return;
  int n = (int)(i >> 2), h = (int)(i & 3);
  float s = 0.f, d = 0.f;
#pragma unroll
  for (int c = 0; c < 16; ++c) {
    float v = hp[(size_t)n * 64 + h * 16 + c];
    s += v * asrc[h * 16 + c];
    d += v * adst[h * 16 + c];
  }
  es[i] = s; ed[i] = d;
}

__global__ void k_edge_max(const int* __restrict__ ei, int E, int N,
                           const float* __restrict__ es, const float* __restrict__ ed,
                           unsigned* __restrict__ smax) {
  long i = (long)blockIdx.x * blockDim.x + threadIdx.x;
  long tot = ((long)E + N) * 4;
  if (i >= tot) return;
  int h = (int)(i & 3); long e = i >> 2;
  int s, d; edge_sd(ei, e, E, s, d);
  float v = es[(size_t)s * 4 + h] + ed[(size_t)d * 4 + h];
  v = v > 0.f ? v : 0.2f * v;
  atomicMax(&smax[(size_t)d * 4 + h], f_flip(v));
}

__global__ void k_edge_agg(const int* __restrict__ ei, int E, int N,
                           const float* __restrict__ es, const float* __restrict__ ed,
                           const unsigned* __restrict__ smax, float* __restrict__ ssum,
                           const float* __restrict__ hp, float* __restrict__ g) {
  long i = (long)blockIdx.x * blockDim.x + threadIdx.x;
  long tot = ((long)E + N) * 4;
  if (i >= tot) return;
  int h = (int)(i & 3); long e = i >> 2;
  int s, d; edge_sd(ei, e, E, s, d);
  float v = es[(size_t)s * 4 + h] + ed[(size_t)d * 4 + h];
  v = v > 0.f ? v : 0.2f * v;
  float ev = __expf(v - f_unflip(smax[(size_t)d * 4 + h]));
  atomicAdd(&ssum[(size_t)d * 4 + h], ev);
  const float* hs = hp + (size_t)s * 64 + h * 16;
  float* gd = g + (size_t)d * 64 + h * 16;
#pragma unroll
  for (int c = 0; c < 16; ++c) atomicAdd(&gd[c], ev * hs[c]);
}

__global__ void k_gat_norm(float* __restrict__ g, const float* __restrict__ ssum, int N) {
  long i = (long)blockIdx.x * blockDim.x + threadIdx.x;
  if (i >= (long)N * 64) return;
  int n = (int)(i >> 6), h = (int)((i >> 4) & 3);
  g[i] /= ssum[(size_t)n * 4 + h];
}

__global__ void k_add2(float* __restrict__ c, const float* __restrict__ a,
                       const float* __restrict__ b, long n) {
  long i = (long)blockIdx.x * blockDim.x + threadIdx.x;
  if (i < n) c[i] = a[i] + b[i];
}

__global__ void k_copy(float* __restrict__ d, const float* __restrict__ s, long n) {
  long i = (long)blockIdx.x * blockDim.x + threadIdx.x;
  if (i < n) d[i] = s[i];
}

// -------------------------------------------------------------- batch norm
__global__ void k_bn_stats(const float* __restrict__ x, float* __restrict__ sums,
                           float* __restrict__ sumsq, int N, int C) {
  int c = threadIdx.x;
  if (c >= C) return;
  int r0 = blockIdx.x * 256;
  int r1 = r0 + 256; if (r1 > N) r1 = N;
  float s = 0.f, q = 0.f;
  for (int r = r0; r < r1; ++r) { float v = x[(size_t)r * C + c]; s += v; q += v * v; }
  atomicAdd(&sums[c], s);
  atomicAdd(&sumsq[c], q);
}

__global__ void k_bn_apply(float* __restrict__ x, const float* __restrict__ sums,
                           const float* __restrict__ sumsq, const float* __restrict__ g,
                           const float* __restrict__ b, int N, int C) {
  long i = (long)blockIdx.x * blockDim.x + threadIdx.x;
  if (i >= (long)N * C) return;
  int c = (int)(i % C);
  float invN = 1.f / (float)N;
  float mu  = sums[c] * invN;
  float var = sumsq[c] * invN - mu * mu;
  float sc  = g[c] * rsqrtf(var + 1e-5f);
  x[i] = sc * (x[i] - mu) + b[c];
}

// ------------------------------------------------------------------ APPNP
__global__ void k_deg(const int* __restrict__ ei, int E, int N, float* __restrict__ deg) {
  long e = (long)blockIdx.x * blockDim.x + threadIdx.x;
  if (e >= (long)E + N) return;
  int s, d; edge_sd(ei, e, E, s, d);
  atomicAdd(&deg[d], 1.f);
}
__global__ void k_dinv(float* __restrict__ deg, int N) {
  int n = blockIdx.x * blockDim.x + threadIdx.x;
  if (n < N) deg[n] = rsqrtf(fmaxf(deg[n], 1.f));
}
__global__ void k_appnp_edge(const int* __restrict__ ei, int E, int N,
                             const float* __restrict__ dinv, const float* __restrict__ h,
                             float* __restrict__ tmp) {
  long i = (long)blockIdx.x * blockDim.x + threadIdx.x;
  long tot = ((long)E + N) * 4;
  if (i >= tot) return;
  int q = (int)(i & 3); long e = i >> 2;
  int s, d; edge_sd(ei, e, E, s, d);
  float w = dinv[s] * dinv[d];
  const float* hs = h + (size_t)s * 64 + q * 16;
  float* td = tmp + (size_t)d * 64 + q * 16;
#pragma unroll
  for (int c = 0; c < 16; ++c) atomicAdd(&td[c], w * hs[c]);
}
__global__ void k_appnp_comb(float* __restrict__ h, const float* __restrict__ tmp,
                             const float* __restrict__ h0, float alpha, long n) {
  long i = (long)blockIdx.x * blockDim.x + threadIdx.x;
  if (i < n) h[i] = (1.f - alpha) * tmp[i] + alpha * h0[i];
}

// ----------------------------------------------------------------- pooling
// h68 padded: stride 96, cols 68..95 zero
__global__ void k_build_h68(const float* __restrict__ h, const float* __restrict__ sz,
                            const float* __restrict__ J, const float* __restrict__ sav,
                            const float* __restrict__ inf, float* __restrict__ h68, int N) {
  long i = (long)blockIdx.x * blockDim.x + threadIdx.x;
  if (i >= (long)N * 96) return;
  int n = (int)(i / 96), j = (int)(i % 96);
  float v = 0.f;
  if (j < 64)       v = h[(size_t)n * 64 + j];
  else if (j == 64) v = sz[n];
  else if (j == 65) v = J[n];
  else if (j == 66) v = sav[n];
  else if (j == 67) v = inf[n];
  h68[i] = v;
}

__global__ void k_dot(const float* __restrict__ X, int K, const float* __restrict__ w,
                      const float* __restrict__ bptr, float* __restrict__ out, int N, int act) {
  int n = blockIdx.x * blockDim.x + threadIdx.x;
  if (n >= N) return;
  float s = bptr ? bptr[0] : 0.f;
  for (int k = 0; k < K; ++k) s += X[(size_t)n * K + k] * w[k];
  if (act == 3) s = 1.f / (1.f + __expf(-s));
  out[n] = s;
}

__global__ void k_pool_max(const float* __restrict__ gate, const int* __restrict__ batch,
                           unsigned* __restrict__ pmax, int N) {
  int n = blockIdx.x * blockDim.x + threadIdx.x;
  if (n < N) atomicMax(&pmax[batch[n]], f_flip(gate[n]));
}
__global__ void k_pool_expsum(float* __restrict__ gate, const int* __restrict__ batch,
                              const unsigned* __restrict__ pmax, float* __restrict__ psum, int N) {
  int n = blockIdx.x * blockDim.x + threadIdx.x;
  if (n >= N) return;
  float e = __expf(gate[n] - f_unflip(pmax[batch[n]]));
  gate[n] = e;
  atomicAdd(&psum[batch[n]], e);
}
__global__ void k_pool_agg(const float* __restrict__ gate, const float* __restrict__ psum,
                           const float* __restrict__ v, const int* __restrict__ batch,
                           float* __restrict__ ctx, int p, int N) {
  long i = (long)blockIdx.x * blockDim.x + threadIdx.x;
  if (i >= (long)N * 4) return;
  int n = (int)(i >> 2), q = (int)(i & 3);
  int b = batch[n];
  float a = gate[n] / psum[b];
  const float* vs = v + (size_t)n * 64 + q * 16;
  float* cd = ctx + (size_t)b * kDCTX + p * 64 + q * 16;
#pragma unroll
  for (int c = 0; c < 16; ++c) atomicAdd(&cd[c], a * vs[c]);
}

__global__ void k_ctx_scalars(float* __restrict__ ctx, const float* nn, const float* om,
                              const float* ph, const float* la, const float* omn,
                              const float* phn, const float* lan, int B) {
  int b = threadIdx.x;
  if (b >= B) return;
  float* c = ctx + (size_t)b * kDCTX + 128;
  c[0] = nn[b]; c[1] = om[b]; c[2] = ph[b]; c[3] = la[b];
  c[4] = omn[b]; c[5] = phn[b]; c[6] = lan[b];
}

// xs padded: stride 224, cols 203..223 zero
__global__ void k_build_xs(const float* __restrict__ h68, const float* __restrict__ ctx,
                           const int* __restrict__ batch, float* __restrict__ xs, int N) {
  long i = (long)blockIdx.x * blockDim.x + threadIdx.x;
  if (i >= (long)N * 224) return;
  int n = (int)(i / 224), j = (int)(i % 224);
  float v = 0.f;
  if (j < 68)       v = h68[(size_t)n * 96 + j];
  else if (j < 203) v = ctx[(size_t)batch[n] * kDCTX + (j - 68)];
  xs[i] = v;
}

__global__ void k_out(const float* __restrict__ s3, const int* __restrict__ batch,
                      float* __restrict__ out, int N) {
  int n = blockIdx.x * blockDim.x + threadIdx.x;
  if (n < N) atomicAdd(&out[batch[n]], s3[n]);
}

// --------------------------------------------------------------- host side
static inline long cdl(long a, long b) { return (a + b - 1) / b; }

template <int NKT>
static void gemm_nkt(dim3 grid, hipStream_t st, const float* A, int lda, int M,
                     const bf16* W, int Nout, const float* bias, float* C, int act) {
  if (!bias) {  // all no-bias call sites use act==0
    k_gemm_bf16<NKT, 0, false><<<grid, 256, 0, st>>>(A, lda, M, W, Nout, bias, C);
  } else if (act == 1) {
    k_gemm_bf16<NKT, 1, true><<<grid, 256, 0, st>>>(A, lda, M, W, Nout, bias, C);
  } else if (act == 2) {
    k_gemm_bf16<NKT, 2, true><<<grid, 256, 0, st>>>(A, lda, M, W, Nout, bias, C);
  } else {
    k_gemm_bf16<NKT, 0, true><<<grid, 256, 0, st>>>(A, lda, M, W, Nout, bias, C);
  }
}

static void gemm(hipStream_t st, const float* A, int lda, int M,
                 const bf16* W, int Kpad, int Nout, const float* bias,
                 float* C, int act) {
  dim3 grid((unsigned)(Nout / 16), (unsigned)cdl(cdl(M, 16), 8));
  switch (Kpad >> 5) {
    case 1:  gemm_nkt<1>(grid, st, A, lda, M, W, Nout, bias, C, act); break;
    case 2:  gemm_nkt<2>(grid, st, A, lda, M, W, Nout, bias, C, act); break;
    case 3:  gemm_nkt<3>(grid, st, A, lda, M, W, Nout, bias, C, act); break;
    case 4:  gemm_nkt<4>(grid, st, A, lda, M, W, Nout, bias, C, act); break;
    default: gemm_nkt<7>(grid, st, A, lda, M, W, Nout, bias, C, act); break;
  }
}

extern "C" void kernel_launch(void* const* d_in, const int* in_sizes, int n_in,
                              void* d_out, int out_size, void* d_ws, size_t ws_size,
                              hipStream_t stream) {
  (void)in_sizes; (void)n_in; (void)out_size; (void)ws_size;
  const float* x     = (const float*)d_in[0];
  const int*   ei    = (const int*)d_in[1];
  const int*   batch = (const int*)d_in[2];
  const float* nno   = (const float*)d_in[3];
  const float* Om    = (const float*)d_in[4];
  const float* Ph    = (const float*)d_in[5];
  const float* La    = (const float*)d_in[6];
  const float* Omn   = (const float*)d_in[7];
  const float* Phn   = (const float*)d_in[8];
  const float* Lan   = (const float*)d_in[9];
  const float* Jv    = (const float*)d_in[10];
  const float* sav   = (const float*)d_in[11];
  const float* infc  = (const float*)d_in[12];
  const float* szc   = (const float*)d_in[13];
  const float* lin0_W = (const float*)d_in[14];
  const float* lin0_b = (const float*)d_in[15];
  const float* gat_W  = (const float*)d_in[16];
  const float* gat_as = (const float*)d_in[17];
  const float* gat_ad = (const float*)d_in[18];
  const float* l1_W   = (const float*)d_in[19];
  const float* bn1_g  = (const float*)d_in[20];
  const float* bn1_b  = (const float*)d_in[21];
  const float* l2_W   = (const float*)d_in[22];
  const float* l2_b   = (const float*)d_in[23];
  const float* l3_W   = (const float*)d_in[24];
  const float* l3_b   = (const float*)d_in[25];
  const float* bn2_g  = (const float*)d_in[26];
  const float* bn2_b  = (const float*)d_in[27];
  const float* g1_W   = (const float*)d_in[28];
  const float* g1_b   = (const float*)d_in[29];
  const float* g2_W   = (const float*)d_in[30];
  const float* g2_b   = (const float*)d_in[31];
  const float* n1_W   = (const float*)d_in[32];
  const float* n1_b   = (const float*)d_in[33];
  const float* n2_W   = (const float*)d_in[34];
  const float* n2_b   = (const float*)d_in[35];
  const float* h1_W   = (const float*)d_in[36];
  const float* h1_b   = (const float*)d_in[37];
  const float* hbn1_g = (const float*)d_in[38];
  const float* hbn1_b = (const float*)d_in[39];
  const float* h2_W   = (const float*)d_in[40];
  const float* h2_b   = (const float*)d_in[41];
  const float* hbn2_g = (const float*)d_in[42];
  const float* hbn2_b = (const float*)d_in[43];
  const float* h3_W   = (const float*)d_in[44];
  const float* h3_b   = (const float*)d_in[45];

  // ---- workspace: packed bf16 weight arena first, fp32 buffers after
  bf16* ar = (bf16*)d_ws;
  size_t ao = 0;
  auto ba = [&](size_t n) -> bf16* { bf16* p = ar + ao; ao += (n + 63) & ~(size_t)63; return p; };
  bf16* wb_lin0   = ba(32 * 64);
  bf16* wb_gat[2] = { ba(64 * 64), ba(64 * 64) };
  bf16* wb_l1e[2] = { ba(64 * 64), ba(64 * 64) };
  bf16* wb_l2[2]  = { ba(64 * 128), ba(64 * 128) };
  bf16* wb_l3[2]  = { ba(128 * 64), ba(128 * 64) };
  bf16* wb_g1[2]  = { ba(96 * 128), ba(96 * 128) };
  bf16* wb_n1[2]  = { ba(96 * 128), ba(96 * 128) };
  bf16* wb_n2[2]  = { ba(128 * 64), ba(128 * 64) };
  bf16* wb_h1 = ba(224 * 128);
  bf16* wb_h2 = ba(128 * 64);

  float* fb = (float*)((char*)d_ws + ((ao * sizeof(bf16) + 255) & ~(size_t)255));
  size_t fo = 0;
  auto fa = [&](size_t n) -> float* { float* p = fb + fo; fo += (n + 63) & ~(size_t)63; return p; };
  float* h    = fa((size_t)kN * 64);
  float* t64a = fa((size_t)kN * 64);   // hp / l3-out / appnp tmp / head s2
  float* t64b = fa((size_t)kN * 64);   // g  / pool value
  float* t128 = fa((size_t)kN * 128);
  float* h0   = fa((size_t)kN * 64);
  float* h68  = fa((size_t)kN * 96);   // padded stride 96
  float* xs   = fa((size_t)kN * 224);  // padded stride 224
  float* es   = fa((size_t)kN * 4);
  float* ed   = fa((size_t)kN * 4);
  float* smax = fa((size_t)kN * 4);    // reinterpreted as unsigned
  float* ssum = fa((size_t)kN * 4);
  float* deg  = fa(kN);
  float* gv   = fa(kN);                // gate score / head s3
  float* xin  = fa((size_t)kN * 32);   // padded stride 32
  float* ctx  = fa((size_t)kB * kDCTX);
  float* pmax = fa(kB);
  float* psum = fa(kB);
  float* bns  = fa(128);
  float* bnq  = fa(128);

  auto fill = [&](float* p, float v, long n) {
    k_fill_f<<<(unsigned)cdl(n, 256), 256, 0, stream>>>(p, v, n);
  };
  auto conv = [&](const float* src, bf16* dst, int K, int Nout, int Kpad, int kmod) {
    long tot = (long)Kpad * Nout;
    k_pack_w<<<(unsigned)cdl(tot, 256), 256, 0, stream>>>(src, dst, K, Nout, Kpad, kmod);
  };
  auto bn = [&](float* xbuf, int n, int Cc, const float* g_, const float* b_) {
    fill(bns, 0.f, 128);
    fill(bnq, 0.f, 128);
    k_bn_stats<<<(unsigned)cdl(n, 256), Cc, 0, stream>>>(xbuf, bns, bnq, n, Cc);
    k_bn_apply<<<(unsigned)cdl((long)n * Cc, 256), 256, 0, stream>>>(xbuf, bns, bnq, g_, b_, n, Cc);
  };

  // ---- pack weights into WMMA fragment order (every launch; no cached state)
  conv(lin0_W, wb_lin0, 8, 64, 32, 8);
  for (int k = 0; k < 2; ++k) {
    conv(gat_W + (size_t)k * 64 * 64, wb_gat[k], 64, 64, 64, 64);
    conv(l1_W + (size_t)k * 16 * 64, wb_l1e[k], 64, 64, 64, 16);  // replicate 16 rows x4 heads
    conv(l2_W + (size_t)k * 64 * 128, wb_l2[k], 64, 128, 64, 64);
    conv(l3_W + (size_t)k * 128 * 64, wb_l3[k], 128, 64, 128, 128);
  }
  for (int p = 0; p < 2; ++p) {
    conv(g1_W + (size_t)p * 68 * 128, wb_g1[p], 68, 128, 96, 68);
    conv(n1_W + (size_t)p * 68 * 128, wb_n1[p], 68, 128, 96, 68);
    conv(n2_W + (size_t)p * 128 * 64, wb_n2[p], 128, 64, 128, 128);
  }
  conv(h1_W, wb_h1, 203, 128, 224, 203);
  conv(h2_W, wb_h2, 128, 64, 128, 128);

  // ---- NodeEncoder: lin0
  k_build_xin<<<(unsigned)cdl((long)kN * 32, 256), 256, 0, stream>>>(x, Jv, szc, xin, kN);
  gemm(stream, xin, 32, kN, wb_lin0, 32, 64, lin0_b, h, 0);

  const long EL = (long)kE + kN;

  // ---- 2 GAT attention layers
  for (int k = 0; k < 2; ++k) {
    gemm(stream, h, 64, kN, wb_gat[k], 64, 64, (const float*)nullptr, t64a, 0);
    k_esed<<<(unsigned)cdl((long)kN * 4, 256), 256, 0, stream>>>(t64a, gat_as + k * 64, gat_ad + k * 64, es, ed, kN);
    fill(smax, 0.f, (long)kN * 4);                       // flipped(-inf) >= 0
    fill(ssum, 0.f, (long)kN * 4);
    fill(t64b, 0.f, (long)kN * 64);
    k_edge_max<<<(unsigned)cdl(EL * 4, 256), 256, 0, stream>>>(ei, kE, kN, es, ed, (unsigned*)smax);
    k_edge_agg<<<(unsigned)cdl(EL * 4, 256), 256, 0, stream>>>(ei, kE, kN, es, ed, (const unsigned*)smax, ssum, t64a, t64b);
    k_gat_norm<<<(unsigned)cdl((long)kN * 64, 256), 256, 0, stream>>>(t64b, ssum, kN);
    gemm(stream, t64b, 64, kN, wb_l1e[k], 64, 64, (const float*)nullptr, t64a, 0);
    k_add2<<<(unsigned)cdl((long)kN * 64, 256), 256, 0, stream>>>(h, h, t64a, (long)kN * 64);
    bn(h, kN, 64, bn1_g + k * 64, bn1_b + k * 64);
    gemm(stream, h, 64, kN, wb_l2[k], 64, 128, l2_b + k * 128, t128, 1);
    gemm(stream, t128, 128, kN, wb_l3[k], 128, 64, l3_b + k * 64, t64a, 0);
    k_add2<<<(unsigned)cdl((long)kN * 64, 256), 256, 0, stream>>>(h, h, t64a, (long)kN * 64);
    bn(h, kN, 64, bn2_g + k * 64, bn2_b + k * 64);
  }

  // ---- APPNP (K=10, alpha=0.1)
  fill(deg, 0.f, kN);
  k_deg<<<(unsigned)cdl(EL, 256), 256, 0, stream>>>(ei, kE, kN, deg);
  k_dinv<<<(unsigned)cdl(kN, 256), 256, 0, stream>>>(deg, kN);
  k_copy<<<(unsigned)cdl((long)kN * 64, 256), 256, 0, stream>>>(h0, h, (long)kN * 64);
  for (int it = 0; it < 10; ++it) {
    fill(t64a, 0.f, (long)kN * 64);
    k_appnp_edge<<<(unsigned)cdl(EL * 4, 256), 256, 0, stream>>>(ei, kE, kN, deg, h, t64a);
    k_appnp_comb<<<(unsigned)cdl((long)kN * 64, 256), 256, 0, stream>>>(h, t64a, h0, 0.1f, (long)kN * 64);
  }

  // ---- features + context pools
  k_build_h68<<<(unsigned)cdl((long)kN * 96, 256), 256, 0, stream>>>(h, szc, Jv, sav, infc, h68, kN);
  fill(ctx, 0.f, (long)kB * kDCTX);
  for (int p = 0; p < 2; ++p) {
    gemm(stream, h68, 96, kN, wb_g1[p], 96, 128, g1_b + p * 128, t128, 1);
    k_dot<<<(unsigned)cdl(kN, 256), 256, 0, stream>>>(t128, 128, g2_W + (size_t)p * 128, g2_b + p, gv, kN, 0);
    gemm(stream, h68, 96, kN, wb_n1[p], 96, 128, n1_b + p * 128, t128, 1);
    gemm(stream, t128, 128, kN, wb_n2[p], 128, 64, n2_b + p * 64, t64b, 0);
    fill(pmax, 0.f, kB);
    fill(psum, 0.f, kB);
    k_pool_max<<<(unsigned)cdl(kN, 256), 256, 0, stream>>>(gv, batch, (unsigned*)pmax, kN);
    k_pool_expsum<<<(unsigned)cdl(kN, 256), 256, 0, stream>>>(gv, batch, (const unsigned*)pmax, psum, kN);
    k_pool_agg<<<(unsigned)cdl((long)kN * 4, 256), 256, 0, stream>>>(gv, psum, t64b, batch, ctx, p, kN);
  }
  k_ctx_scalars<<<1, 64, 0, stream>>>(ctx, nno, Om, Ph, La, Omn, Phn, Lan, kB);

  // ---- scoring head
  k_build_xs<<<(unsigned)cdl((long)kN * 224, 256), 256, 0, stream>>>(h68, ctx, batch, xs, kN);
  gemm(stream, xs, 224, kN, wb_h1, 224, 128, h1_b, t128, 2);
  bn(t128, kN, 128, hbn1_g, hbn1_b);
  gemm(stream, t128, 128, kN, wb_h2, 128, 64, h2_b, t64a, 2);
  bn(t64a, kN, 64, hbn2_g, hbn2_b);
  k_dot<<<(unsigned)cdl(kN, 256), 256, 0, stream>>>(t64a, 64, h3_W, h3_b, gv, kN, 3);

  float* out = (float*)d_out;
  fill(out, 0.f, kB);
  k_out<<<(unsigned)cdl(kN, 256), 256, 0, stream>>>(gv, batch, out, kN);
}
